// HybridAttentionBlock_76776835383418
// MI455X (gfx1250) — compile-verified, matrix operand order checked by reading, not run on hardware
//
#include <hip/hip_runtime.h>
#include <hip/hip_bf16.h>

// ---------------------------------------------------------------------------
// HybridAttentionBlock for MI455X (gfx1250): every GEMM-like phase runs on
// v_wmma_f32_16x16x32_bf16 (fp32 accum); global->LDS staging uses the CDNA5
// async-copy path (global_load_async_to_lds_b128 + s_wait_asynccnt) with
// double buffering; B operands (weights, V) are pre-swizzled into WMMA
// fragment order so each lane's B fragment is two aligned ds_load_b128.
// ---------------------------------------------------------------------------

typedef __attribute__((ext_vector_type(16))) __bf16 bf16x16;
typedef __attribute__((ext_vector_type(8)))  float  f32x8;

#define DIMC 192
#define HH 192
#define WW 192
#define BB 4
#define NWIN 576          // (192/8)^2 windows per image
#define BWIN 2304         // B * NWIN
#define NTOK 64           // tokens per window
#define NHEAD 6
#define HDIM 32
#define MTOK 147456       // total tokens == total pixels
#define HWSZ 36864        // H*W
#define SCALEQ 0.17677669529663687f

// ---- bf16 helpers ---------------------------------------------------------
__device__ __forceinline__ unsigned short f2us(float f) {
  unsigned u = __builtin_bit_cast(unsigned, f);
  unsigned r = u + 0x7FFFu + ((u >> 16) & 1u);
  return (unsigned short)(r >> 16);
}
__device__ __forceinline__ __bf16 us2bf(unsigned short u) {
  return __builtin_bit_cast(__bf16, u);
}
__device__ __forceinline__ float gelu_exact(float x) {
  return 0.5f * x * (1.0f + erff(x * 0.70710678118654752f));
}

// Swizzled B layout: for a KxN matrix (K%32==0, N%16==0), element (k,n) goes
// to tile (kt=k/32, nt=n/16) — tiles n-major within a K-chunk — at position
// lane*16+e inside the 512-element tile, matching the WMMA B fragment order
// (inverse of the ISA 7.12.2 16-bit packing).
__device__ __forceinline__ size_t bswz(int k, int n, int N) {
  int kt = k >> 5, kk = k & 31;
  int nt = n >> 4, n16 = n & 15;
  int half = kk >> 4, rem16 = kk & 15;
  int hi = rem16 >> 3, rem8 = rem16 & 7;
  int e = 2 * ((rem8 >> 1) + half * 4) + (rem8 & 1);
  int lane = hi * 16 + n16;
  return ((size_t)(kt * (N >> 4) + nt) << 9) + lane * 16 + e;
}

// ---- CDNA5 async global->LDS copy (ASYNCcnt-tracked) ----------------------
__device__ __forceinline__ void async_b128(void* lds, const void* g) {
  unsigned l = (unsigned)(unsigned long long)lds;
  unsigned long long a = (unsigned long long)g;
  asm volatile("global_load_async_to_lds_b128 %0, %1, off"
               :: "v"(l), "v"(a) : "memory");
}
__device__ __forceinline__ void wait_async0() {
  asm volatile("s_wait_asynccnt 0x0" ::: "memory");
}
__device__ __forceinline__ void wait_async3() {
  asm volatile("s_wait_asynccnt 0x3" ::: "memory");
}

// ---- WMMA wrappers --------------------------------------------------------
__device__ __forceinline__ f32x8 wmma_bf16(bf16x16 a, bf16x16 b, f32x8 c) {
  return __builtin_amdgcn_wmma_f32_16x16x32_bf16(
      /*neg_a=*/false, a, /*neg_b=*/false, b,
      /*c_mod=*/(short)0, c, /*reuse_a=*/false, /*reuse_b=*/false);
}

// A operand: 16x32 bf16 tile, row-major in LDS. Each lane's 16 values are
// two contiguous 16B runs -> compiler emits 2x ds_load_b128.
__device__ __forceinline__ bf16x16 load_frag_a(const unsigned short* p, int stride) {
  int lane = threadIdx.x & 31;
  const unsigned short* row = p + (lane & 15) * stride;
  int hi = (lane >> 4) * 8;
  bf16x16 f;
#pragma unroll
  for (int j = 0; j < 8; ++j) {
    int kk = ((j < 4) ? (2 * j) : (16 + 2 * (j - 4))) + hi;
    f[2 * j]     = us2bf(row[kk]);
    f[2 * j + 1] = us2bf(row[kk + 1]);
  }
  return f;
}

// B operand from a swizzled 512-element tile: two aligned b128 loads.
struct U32B { uint4 a, b; };
__device__ __forceinline__ bf16x16 load_frag_bs(const unsigned short* tile) {
  int lane = threadIdx.x & 31;
  const uint4* p = reinterpret_cast<const uint4*>(tile + lane * 16);
  U32B t;
  t.a = p[0];
  t.b = p[1];
  return __builtin_bit_cast(bf16x16, t);
}

// ---- weight prep ----------------------------------------------------------
// src: (Nfeat, Kdim) row-major fp32 (out-feature major, as in the reference
// weights). dst: bf16 swizzled-B layout for the Kdim x Nfeat GEMM operand.
__global__ __launch_bounds__(256) void makeb_bf16_kernel(
    const float* __restrict__ src, unsigned short* __restrict__ dst,
    int Nfeat, int Kdim) {
  int idx = blockIdx.x * 256 + threadIdx.x;
  if (idx >= Nfeat * Kdim) return;
  int n = idx / Kdim, k = idx % Kdim;
  dst[bswz(k, n, Nfeat)] = f2us(src[idx]);
}

// conv weight (O=192, I=192, 3, 3) -> per-tap swizzled-B (K=ci 192, N=o 192)
__global__ __launch_bounds__(256) void convw_bf16_kernel(
    const float* __restrict__ src, unsigned short* __restrict__ dst) {
  int idx = blockIdx.x * 256 + threadIdx.x;  // < 331776
  int o = idx / 1728;
  int r = idx % 1728;
  int ci = r / 9, tap = r % 9;
  dst[(size_t)tap * (DIMC * DIMC) + bswz(ci, o, DIMC)] = f2us(src[idx]);
}

// ---- LN1 + roll(-4,-4) + window partition -> token-major bf16 -------------
__global__ __launch_bounds__(256) void ln1_shift_part_kernel(
    const float* __restrict__ x, const float* __restrict__ g,
    const float* __restrict__ bt, unsigned short* __restrict__ out) {
  int t = blockIdx.x * 256 + threadIdx.x;      // token 0..147455
  int b_ = t >> 6, n = t & 63;
  int b = b_ / NWIN, wi = b_ % NWIN;
  int wh = wi / 24, wwq = wi % 24;
  int pr = n >> 3, pc = n & 7;
  int hh = wh * 8 + pr + 4; if (hh >= HH) hh -= HH;   // roll(-4) => read +4
  int wp = wwq * 8 + pc + 4; if (wp >= WW) wp -= WW;
  const float* px = x + (size_t)b * DIMC * HWSZ + hh * WW + wp;
  float s = 0.f, ss = 0.f;
  for (int c = 0; c < DIMC; ++c) {
    float v = px[(size_t)c * HWSZ];
    s += v; ss += v * v;
  }
  float mu = s * (1.f / DIMC);
  float var = ss * (1.f / DIMC) - mu * mu;
  float inv = rsqrtf(var + 1e-5f);
  unsigned short* po = out + (size_t)t * DIMC;
  for (int c = 0; c < DIMC; ++c) {
    float v = (px[(size_t)c * HWSZ] - mu) * inv * g[c] + bt[c];
    po[c] = f2us(v);
  }
}

// ---- LN2 (pixel-major) ----------------------------------------------------
__global__ __launch_bounds__(256) void ln2_kernel(
    const float* __restrict__ x, const float* __restrict__ g,
    const float* __restrict__ bt, unsigned short* __restrict__ out) {
  int p = blockIdx.x * 256 + threadIdx.x;      // pixel 0..147455 (b,h,w)
  int b = p / HWSZ, hw = p % HWSZ;
  const float* px = x + (size_t)b * DIMC * HWSZ + hw;
  float s = 0.f, ss = 0.f;
  for (int c = 0; c < DIMC; ++c) {
    float v = px[(size_t)c * HWSZ];
    s += v; ss += v * v;
  }
  float mu = s * (1.f / DIMC);
  float var = ss * (1.f / DIMC) - mu * mu;
  float inv = rsqrtf(var + 1e-5f);
  unsigned short* po = out + (size_t)p * DIMC;
  for (int c = 0; c < DIMC; ++c) {
    float v = (px[(size_t)c * HWSZ] - mu) * inv * g[c] + bt[c];
    po[c] = f2us(v);
  }
}

// ---- generic bf16 WMMA GEMM: C = A(MxK) * B(KxN), B swizzled --------------
// Double-buffered async global->LDS pipeline (ASYNCcnt) overlapped with WMMA.
// mode 0: QKV  (bias, scale q, scatter bf16; V goes to swizzled tile layout)
// mode 1: MLP1 (bias + GELU -> bf16 out0, row-major MxN)
// mode 2: MLP2 (bias + residual(resid) -> fp32 NCHW outf)
__global__ __launch_bounds__(256) void gemm_bf16_kernel(
    const unsigned short* __restrict__ A, const unsigned short* __restrict__ Bm,
    int M, int N, int K, int mode, const float* __restrict__ bias,
    unsigned short* __restrict__ out0, unsigned short* __restrict__ out1,
    unsigned short* __restrict__ out2, const float* __restrict__ resid,
    float* __restrict__ outf) {
  __shared__ __align__(16) unsigned short As[2][128][40];
  __shared__ __align__(16) unsigned short Bsw[2][2048];   // 4 swizzled tiles
  int tid = threadIdx.x;
  int wid = tid >> 5, lane = tid & 31;
  int mw = wid & 3, nw = wid >> 2;          // wave grid 4x2 -> 32x32 per wave
  int m0 = blockIdx.y * 128;
  int n0 = blockIdx.x * 64;
  int arow = tid >> 2, ag = tid & 3;        // A: rows arow and arow+64
  int ntiles = N >> 4;
  f32x8 acc[2][2] = {};

  auto issue_tile = [&](int buf, int k0) {
    async_b128(&As[buf][arow][ag * 8],
               A + (size_t)(m0 + arow) * K + k0 + ag * 8);
    async_b128(&As[buf][arow + 64][ag * 8],
               A + (size_t)(m0 + arow + 64) * K + k0 + ag * 8);
    // 4 contiguous swizzled tiles for cols n0..n0+63 at K-chunk k0
    size_t bbase = ((size_t)((k0 >> 5) * ntiles + (n0 >> 4)) << 9);
    async_b128(&Bsw[buf][tid * 8], Bm + bbase + tid * 8);
  };

  issue_tile(0, 0);
  int nk = K >> 5;
  for (int i = 0; i < nk; ++i) {
    int cur = i & 1;
    if (i + 1 < nk) {
      issue_tile(cur ^ 1, (i + 1) << 5);
      wait_async3();                         // current tile's 3 copies landed
    } else {
      wait_async0();
    }
    __syncthreads();
    bf16x16 bf0 = load_frag_bs(&Bsw[cur][(nw * 2 + 0) << 9]);
    bf16x16 bf1 = load_frag_bs(&Bsw[cur][(nw * 2 + 1) << 9]);
#pragma unroll
    for (int mt = 0; mt < 2; ++mt) {
      bf16x16 af = load_frag_a(&As[cur][mw * 32 + mt * 16][0], 40);
      acc[mt][0] = wmma_bf16(af, bf0, acc[mt][0]);
      acc[mt][1] = wmma_bf16(af, bf1, acc[mt][1]);
    }
    __syncthreads();                         // protect cur from next overwrite
  }
  int ncol = lane & 15;
  int moff = (lane >> 4) * 8;
#pragma unroll
  for (int mt = 0; mt < 2; ++mt)
#pragma unroll
    for (int nt = 0; nt < 2; ++nt)
#pragma unroll
      for (int j = 0; j < 8; ++j) {
        int gm = m0 + mw * 32 + mt * 16 + j + moff;
        int gn = n0 + nw * 32 + nt * 16 + ncol;
        float v = acc[mt][nt][j] + bias[gn];
        if (mode == 0) {
          int which = gn / DIMC;
          int hd_ = gn % DIMC;
          int head = hd_ >> 5, d = hd_ & 31;
          int bh = (gm >> 6) * NHEAD + head;
          int tok = gm & 63;
          if (which == 0) {
            v *= SCALEQ;
            out0[((size_t)bh * NTOK + tok) * HDIM + d] = f2us(v);
          } else if (which == 1) {
            out1[((size_t)bh * NTOK + tok) * HDIM + d] = f2us(v);
          } else {
            // V in per-(window,head) swizzled-B layout (K=64 tokens, N=32)
            out2[(size_t)bh * 2048 + bswz(tok, d, HDIM)] = f2us(v);
          }
        } else if (mode == 1) {
          out0[(size_t)gm * 768 + gn] = f2us(gelu_exact(v));
        } else {
          int b = gm / HWSZ, hw = gm % HWSZ;
          size_t idx = ((size_t)b * DIMC + gn) * HWSZ + hw;
          outf[idx] = resid[idx] + v;
        }
      }
}

// ---- fused windowed attention --------------------------------------------
// One block (128 thr / 4 waves) per (window, head). Q/K/V staged via async
// LDS copies (V pre-swizzled). Bias & shift-mask computed analytically.
// Epilogue fuses window-reverse + roll(+4) + residual.
__global__ __launch_bounds__(128) void attn_kernel(
    const unsigned short* __restrict__ Q, const unsigned short* __restrict__ Kmat,
    const unsigned short* __restrict__ V, const float* __restrict__ rel_table,
    const float* __restrict__ xin, float* __restrict__ x1out) {
  __shared__ __align__(16) unsigned short Qs[64][40];
  __shared__ __align__(16) unsigned short Ks[64][40];
  __shared__ __align__(16) unsigned short Vsw[2048];   // 4 swizzled tiles
  __shared__ __align__(16) unsigned short Ps[64][72];
  int bh = blockIdx.x;
  int head = bh % NHEAD;
  int b_ = bh / NHEAD;
  int b = b_ / NWIN;
  int wi = b_ % NWIN;
  int wh = wi / 24, wwq = wi % 24;
  int tid = threadIdx.x, wid = tid >> 5, lane = tid & 31;
  size_t base = (size_t)bh * NTOK * HDIM;   // 2048 elements per (win,head)
#pragma unroll
  for (int r = 0; r < 2; ++r) {             // 64x32 each, 8-elem vecs
    int e = tid + r * 128;
    int row = e >> 2, gq = e & 3;
    async_b128(&Qs[row][gq * 8], Q + base + row * HDIM + gq * 8);
    async_b128(&Ks[row][gq * 8], Kmat + base + row * HDIM + gq * 8);
    async_b128(&Vsw[e * 8], V + base + e * 8);       // already swizzled
  }
  wait_async0();
  __syncthreads();
  int rbase = wid * 16;
  // S = Q * K^T  (B operand columns == K rows -> reuse the A row-loader)
  f32x8 s[4] = {};
  bf16x16 aq = load_frag_a(&Qs[rbase][0], 40);
#pragma unroll
  for (int nb = 0; nb < 4; ++nb) {
    bf16x16 bk = load_frag_a(&Ks[nb * 16][0], 40);
    s[nb] = wmma_bf16(aq, bk, s[nb]);
  }
  int ncol = lane & 15;
  int moff = (lane >> 4) * 8;
#pragma unroll
  for (int j = 0; j < 8; ++j) {
    int t1 = rbase + j + moff;               // query token
    int pr1 = t1 >> 3, pc1 = t1 & 7;
    int h1 = wh * 8 + pr1, w1 = wwq * 8 + pc1;
    int id1 = (h1 < 184 ? 0 : (h1 < 188 ? 1 : 2)) * 3 +
              (w1 < 184 ? 0 : (w1 < 188 ? 1 : 2));
    float mx = -1e30f;
#pragma unroll
    for (int nb = 0; nb < 4; ++nb) {
      int t2 = nb * 16 + ncol;               // key token
      int pr2 = t2 >> 3, pc2 = t2 & 7;
      int h2 = wh * 8 + pr2, w2 = wwq * 8 + pc2;
      int id2 = (h2 < 184 ? 0 : (h2 < 188 ? 1 : 2)) * 3 +
                (w2 < 184 ? 0 : (w2 < 188 ? 1 : 2));
      int ridx = (pr1 - pr2 + 7) * 15 + (pc1 - pc2 + 7);
      float v = s[nb][j] + rel_table[ridx * NHEAD + head] +
                (id1 != id2 ? -100.0f : 0.0f);
      s[nb][j] = v;
      mx = fmaxf(mx, v);
    }
#pragma unroll
    for (int o = 8; o >= 1; o >>= 1) mx = fmaxf(mx, __shfl_xor(mx, o, 32));
    float ssum = 0.f;
#pragma unroll
    for (int nb = 0; nb < 4; ++nb) {
      float e = __expf(s[nb][j] - mx);
      s[nb][j] = e;
      ssum += e;
    }
#pragma unroll
    for (int o = 8; o >= 1; o >>= 1) ssum += __shfl_xor(ssum, o, 32);
    float inv = 1.0f / ssum;
#pragma unroll
    for (int nb = 0; nb < 4; ++nb)
      Ps[t1][nb * 16 + ncol] = f2us(s[nb][j] * inv);
  }
  __syncthreads();
  // O = P * V   (V tiles: (kb,cb) -> tile kb*2+cb)
  f32x8 oacc[2] = {};
#pragma unroll
  for (int kb = 0; kb < 2; ++kb) {
    bf16x16 ap = load_frag_a(&Ps[rbase][kb * 32], 72);
    bf16x16 bv0 = load_frag_bs(&Vsw[(kb * 2 + 0) << 9]);
    bf16x16 bv1 = load_frag_bs(&Vsw[(kb * 2 + 1) << 9]);
    oacc[0] = wmma_bf16(ap, bv0, oacc[0]);
    oacc[1] = wmma_bf16(ap, bv1, oacc[1]);
  }
#pragma unroll
  for (int cb = 0; cb < 2; ++cb)
#pragma unroll
    for (int j = 0; j < 8; ++j) {
      int t = rbase + j + moff;
      int d = cb * 16 + ncol;
      int ch = head * HDIM + d;
      int pr = t >> 3, pc = t & 7;
      int hg = wh * 8 + pr + 4; if (hg >= HH) hg -= HH;  // roll back (+4)
      int wg = wwq * 8 + pc + 4; if (wg >= WW) wg -= WW;
      size_t idx = ((size_t)b * DIMC + ch) * HWSZ + hg * WW + wg;
      x1out[idx] = xin[idx] + oacc[cb][j];
    }
}

// ---- depthwise 3x3 gate: x2 = x1 + dw(x1) + gb; also emit bf16 NHWC -------
__global__ __launch_bounds__(256) void gate_kernel(
    const float* __restrict__ x1, const float* __restrict__ gw,
    const float* __restrict__ gb, float* __restrict__ x2f,
    unsigned short* __restrict__ x2bf) {
  size_t n = (size_t)blockIdx.x * 256 + threadIdx.x;   // NCHW flat
  int w = (int)(n % WW);
  int h = (int)((n / WW) % HH);
  int c = (int)((n / HWSZ) % DIMC);
  int b = (int)(n / ((size_t)DIMC * HWSZ));
  const float* pl = x1 + ((size_t)b * DIMC + c) * HWSZ;
  float s = gb[c];
#pragma unroll
  for (int kh = 0; kh < 3; ++kh) {
    int hh = h + kh - 1;
    if (hh < 0 || hh >= HH) continue;
#pragma unroll
    for (int kw = 0; kw < 3; ++kw) {
      int wp = w + kw - 1;
      if (wp < 0 || wp >= WW) continue;
      s += pl[hh * WW + wp] * gw[c * 9 + kh * 3 + kw];
    }
  }
  float v = x1[n] + s;
  x2f[n] = v;
  x2bf[(((size_t)b * HH + h) * WW + w) * DIMC + c] = f2us(v);
}

// ---- dense 3x3 conv as 9-tap implicit WMMA GEMM ---------------------------
// Block: 64 pixels (one W-strip) x all 192 output channels. 8 waves: 4 M
// blocks x 2 N halves of 96, 6 accumulators per wave. Swizzled weight slices
// stream in via async LDS copies; boundary-padded activations use VGPR path.
// mode 0: bias+GELU -> bf16 NHWC; mode 1: bias -> fp32 NCHW.
__global__ __launch_bounds__(256) void conv3x3_kernel(
    const unsigned short* __restrict__ X,     // bf16 NHWC
    const unsigned short* __restrict__ Wt,    // bf16 [tap][swizzled 192x192]
    const float* __restrict__ bias, int mode,
    unsigned short* __restrict__ out_us, float* __restrict__ out_f) {
  __shared__ __align__(16) unsigned short As[64][40];
  __shared__ __align__(16) unsigned short Bsw[6144];    // 12 swizzled tiles
  int blk = blockIdx.x;
  int wt0 = (blk % 3) * 64;
  int h = (blk / 3) % HH;
  int b = blk / (3 * HH);
  int tid = threadIdx.x, wid = tid >> 5, lane = tid & 31;
  int mwv = wid & 3;            // 16-row block of pixels
  int nh = wid >> 2;            // 96-col half of output channels
  f32x8 acc[6] = {};
  for (int tap = 0; tap < 9; ++tap) {
    int dy = tap / 3 - 1, dx = tap % 3 - 1;
    int hs = h + dy;
    for (int kc = 0; kc < DIMC; kc += 32) {
      // B: 12 contiguous swizzled tiles for this (tap, K-chunk)
      size_t bbase = (size_t)tap * (DIMC * DIMC) + ((size_t)((kc >> 5) * 12) << 9);
#pragma unroll
      for (int r = 0; r < 3; ++r) {
        int e = tid + r * 256;
        async_b128(&Bsw[e * 8], Wt + bbase + e * 8);
      }
      {                          // A: 64 pixels x 32 in-channels (shifted)
        int i = tid >> 2, gq = tid & 3;
        int wsrc = wt0 + i + dx;
        uint4 val = make_uint4(0u, 0u, 0u, 0u);
        if (hs >= 0 && hs < HH && wsrc >= 0 && wsrc < WW)
          val = *reinterpret_cast<const uint4*>(
              X + (((size_t)b * HH + hs) * WW + wsrc) * DIMC + kc + gq * 8);
        *reinterpret_cast<uint4*>(&As[i][gq * 8]) = val;
      }
      wait_async0();
      __syncthreads();
      bf16x16 af = load_frag_a(&As[mwv * 16][0], 40);
#pragma unroll
      for (int nt = 0; nt < 6; ++nt) {
        bf16x16 bfr = load_frag_bs(&Bsw[(nh * 6 + nt) << 9]);
        acc[nt] = wmma_bf16(af, bfr, acc[nt]);
      }
      __syncthreads();
    }
  }
  int ncol = lane & 15;
  int moff = (lane >> 4) * 8;
#pragma unroll
  for (int nt = 0; nt < 6; ++nt) {
    int o = nh * 96 + nt * 16 + ncol;
    float bia = bias[o];
#pragma unroll
    for (int j = 0; j < 8; ++j) {
      int w = wt0 + mwv * 16 + j + moff;
      float v = acc[nt][j] + bia;
      if (mode == 0) {
        out_us[(((size_t)b * HH + h) * WW + w) * DIMC + o] = f2us(gelu_exact(v));
      } else {
        out_f[((size_t)b * DIMC + o) * HWSZ + h * WW + w] = v;
      }
    }
  }
}

// ---- global average pool per (b,c) ----------------------------------------
__global__ __launch_bounds__(256) void pool_kernel(
    const float* __restrict__ y, float* __restrict__ p) {
  __shared__ float sm[256];
  int bc = blockIdx.x;
  const float* src = y + (size_t)bc * HWSZ;
  float s = 0.f;
  for (int i = threadIdx.x; i < HWSZ; i += 256) s += src[i];
  sm[threadIdx.x] = s;
  __syncthreads();
  for (int o = 128; o > 0; o >>= 1) {
    if (threadIdx.x < (unsigned)o) sm[threadIdx.x] += sm[threadIdx.x + o];
    __syncthreads();
  }
  if (threadIdx.x == 0) p[bc] = sm[0] * (1.0f / HWSZ);
}

// ---- SE MLP (tiny): wgt = sigmoid(relu(p W1^T + b1) W2^T + b2) ------------
__global__ __launch_bounds__(768) void se_kernel(
    const float* __restrict__ p, const float* __restrict__ w1,
    const float* __restrict__ b1, const float* __restrict__ w2,
    const float* __restrict__ b2, float* __restrict__ wgt) {
  __shared__ float hid[BB * 12];
  int t = threadIdx.x;
  if (t < BB * 12) {
    int b = t / 12, i = t % 12;
    float s = b1[i];
    for (int c = 0; c < DIMC; ++c) s += p[b * DIMC + c] * w1[i * DIMC + c];
    hid[t] = fmaxf(s, 0.f);
  }
  __syncthreads();
  if (t < BB * DIMC) {
    int b = t / DIMC, c = t % DIMC;
    float s = b2[c];
#pragma unroll
    for (int i = 0; i < 12; ++i) s += hid[b * 12 + i] * w2[c * 12 + i];
    wgt[t] = 1.0f / (1.0f + __expf(-s));
  }
}

// ---- SE apply: x3 = y2*wgt + x2 (in place over x2) ------------------------
__global__ __launch_bounds__(256) void se_apply_kernel(
    const float* __restrict__ y2, const float* __restrict__ wgt,
    float* __restrict__ x) {
  size_t n = (size_t)blockIdx.x * 256 + threadIdx.x;
  int bc = (int)(n / HWSZ);
  x[n] = y2[n] * wgt[bc] + x[n];
}

// ---------------------------------------------------------------------------
extern "C" void kernel_launch(void* const* d_in, const int* in_sizes, int n_in,
                              void* d_out, int out_size, void* d_ws, size_t ws_size,
                              hipStream_t stream) {
  (void)in_sizes; (void)n_in; (void)out_size; (void)ws_size;
  const float* x       = (const float*)d_in[0];
  const float* n1g     = (const float*)d_in[1];
  const float* n1b     = (const float*)d_in[2];
  const float* qkv_w   = (const float*)d_in[3];
  const float* qkv_b   = (const float*)d_in[4];
  const float* rel_tab = (const float*)d_in[5];
  const float* n2g     = (const float*)d_in[6];
  const float* n2b     = (const float*)d_in[7];
  const float* mlp_w1  = (const float*)d_in[8];
  const float* mlp_b1  = (const float*)d_in[9];
  const float* mlp_w2  = (const float*)d_in[10];
  const float* mlp_b2  = (const float*)d_in[11];
  const float* c1w     = (const float*)d_in[12];
  const float* c1b     = (const float*)d_in[13];
  const float* c2w     = (const float*)d_in[14];
  const float* c2b     = (const float*)d_in[15];
  const float* se_w1   = (const float*)d_in[16];
  const float* se_b1   = (const float*)d_in[17];
  const float* se_w2   = (const float*)d_in[18];
  const float* se_b2   = (const float*)d_in[19];
  const float* gate_w  = (const float*)d_in[20];
  const float* gate_b  = (const float*)d_in[21];
  float* out = (float*)d_out;

  // workspace layout (bytes), all 16B aligned
  char* ws = (char*)d_ws;
  const size_t SZ_R1 = (size_t)MTOK * 768 * 2;   // qkv scatter / MLP hidden (bf16)
  const size_t SZ_R2 = (size_t)MTOK * DIMC * 2;  // bf16 activations (NHWC/token)
  const size_t SZ_R3 = (size_t)MTOK * DIMC * 4;  // fp32 NCHW
  unsigned short* R1  = (unsigned short*)ws;
  unsigned short* R2  = (unsigned short*)(ws + SZ_R1);
  unsigned short* R2b = (unsigned short*)(ws + SZ_R1 + SZ_R2);
  float* R3 = (float*)(ws + SZ_R1 + 2 * SZ_R2);
  float* R4 = (float*)(ws + SZ_R1 + 2 * SZ_R2 + SZ_R3);
  unsigned short* qkvT = (unsigned short*)(ws + SZ_R1 + 2 * SZ_R2 + 2 * SZ_R3);
  unsigned short* wc1  = qkvT + 576 * 192;
  unsigned short* wc2  = wc1 + 9 * 192 * 192;
  unsigned short* w1T  = wc2 + 9 * 192 * 192;
  unsigned short* w2T  = w1T + 768 * 192;
  float* pbuf = (float*)(w2T + 192 * 768);
  float* wgt  = pbuf + BB * DIMC;

  unsigned short* qp = R1;
  unsigned short* kp = R1 + (size_t)MTOK * DIMC;
  unsigned short* vp = R1 + 2 * (size_t)MTOK * DIMC;

  // 1) weight conversion / swizzling to WMMA B-fragment order
  makeb_bf16_kernel<<<(576 * 192 + 255) / 256, 256, 0, stream>>>(qkv_w, qkvT, 576, 192);
  makeb_bf16_kernel<<<(768 * 192 + 255) / 256, 256, 0, stream>>>(mlp_w1, w1T, 768, 192);
  makeb_bf16_kernel<<<(192 * 768 + 255) / 256, 256, 0, stream>>>(mlp_w2, w2T, 192, 768);
  convw_bf16_kernel<<<331776 / 256, 256, 0, stream>>>(c1w, wc1);
  convw_bf16_kernel<<<331776 / 256, 256, 0, stream>>>(c2w, wc2);

  // 2) LN1 + shift + window partition -> R2 (token-major bf16)
  ln1_shift_part_kernel<<<MTOK / 256, 256, 0, stream>>>(x, n1g, n1b, R2);

  // 3) QKV GEMM (M=147456, N=576, K=192) -> per-head q/k (row) + v (swizzled)
  gemm_bf16_kernel<<<dim3(576 / 64, MTOK / 128), 256, 0, stream>>>(
      R2, qkvT, MTOK, 576, 192, 0, qkv_b, qp, kp, vp, nullptr, nullptr);

  // 4) fused attention + window reverse + roll + residual -> R3 (x1)
  attn_kernel<<<BWIN * NHEAD, 128, 0, stream>>>(qp, kp, vp, rel_tab, x, R3);

  // 5) depthwise gate -> R4 (x2 fp32 NCHW) and R2 (x2 bf16 NHWC)
  gate_kernel<<<(int)((size_t)MTOK * DIMC / 256), 256, 0, stream>>>(
      R3, gate_w, gate_b, R4, R2);

  // 6) conv1 (+GELU) -> R2b (y1 bf16 NHWC); conv2 -> R3 (y2 fp32 NCHW)
  conv3x3_kernel<<<BB * HH * 3, 256, 0, stream>>>(R2, wc1, c1b, 0, R2b, nullptr);
  conv3x3_kernel<<<BB * HH * 3, 256, 0, stream>>>(R2b, wc2, c2b, 1, nullptr, R3);

  // 7) SE: pool y2, tiny MLP, apply: R4 <- y2*wgt + x2  (x3)
  pool_kernel<<<BB * DIMC, 256, 0, stream>>>(R3, pbuf);
  se_kernel<<<1, 768, 0, stream>>>(pbuf, se_w1, se_b1, se_w2, se_b2, wgt);
  se_apply_kernel<<<(int)((size_t)MTOK * DIMC / 256), 256, 0, stream>>>(R3, wgt, R4);

  // 8) MLP: LN2 -> R2; h = GELU(x W1^T + b1) -> R1; out = x3 + h W2^T + b2
  ln2_kernel<<<MTOK / 256, 256, 0, stream>>>(R4, n2g, n2b, R2);
  gemm_bf16_kernel<<<dim3(768 / 64, MTOK / 128), 256, 0, stream>>>(
      R2, w1T, MTOK, 768, 192, 1, mlp_b1, R1, nullptr, nullptr, nullptr, nullptr);
  gemm_bf16_kernel<<<dim3(192 / 64, MTOK / 128), 256, 0, stream>>>(
      R1, w2T, MTOK, 192, 768, 2, mlp_b2, nullptr, nullptr, nullptr, R4, out);
}
